// GatedWaveDynamicsLayer_73169062855019
// MI455X (gfx1250) — compile-verified, matrix operand order checked by reading, not run on hardware
//
#include <hip/hip_runtime.h>

typedef _Float16 half_t;
typedef __attribute__((ext_vector_type(16))) _Float16 v16h;
typedef __attribute__((ext_vector_type(8)))  _Float16 v8h;
typedef __attribute__((ext_vector_type(4)))  _Float16 v4h;
typedef __attribute__((ext_vector_type(8)))  float    v8f;

#define S_     4
#define DIN    1024
#define DD     256
#define B_     4
#define T_     4096
#define NTOK   (B_ * T_)        // 16384 token rows
#define MT     64               // tokens per block tile
#define FR     16               // halves per lane fragment run
#define CHUNK  64
#define NCHUNK (T_ / CHUNK)     // 64
#define BIGN   (S_ * DIN * DD)  // elements per big weight = 1,048,576

__device__ __constant__ float c_r[S_]  = {1.0f, 0.999f, 0.9495f, 0.9f};
__device__ __constant__ float c_th[S_] = {0.0f, 0.01f, 0.505f, 1.0f};
__device__ __constant__ float c_bs[S_] = {0.0625f, 0.001f, 0.0505f, 0.1f};

// Fragment swizzle: 16-bit WMMA operand layout (wave32).
// kk = K index within 32-chunk; col = col (B) or row (A) 0..15.
// lanes 0-15 get K {0..7,16..23}; lanes 16-31 get K {8..15,24..31}.
__device__ __forceinline__ int frag_lane(int kk, int col) {
  return (((kk >> 3) & 1) << 4) | col;
}
__device__ __forceinline__ int frag_elem(int kk) {
  return (kk & 7) + ((kk & 16) >> 1);
}

// Load one 32-byte fragment (16 halves, contiguous per lane).
__device__ __forceinline__ v16h ld_frag(const half_t* p) {
  const v8h lo = *(const v8h*)p;
  const v8h hi = *(const v8h*)(p + 8);
  return __builtin_shufflevector(lo, hi, 0, 1, 2, 3, 4, 5, 6, 7,
                                 8, 9, 10, 11, 12, 13, 14, 15);
}

// ============ K0: one-shot weight convert fp32 -> f16 fragment order =========
// Big weights (K=1024):  dst[w][s][nt4][kc32][fi4][lane32][16]
// Wg2      (K=256):      dst[3][s][nt4][kc8][fi4][lane32][16]
__global__ __launch_bounds__(256)
void k0_prep_weights(const float* __restrict__ Wg1, const float* __restrict__ Wdr,
                     const float* __restrict__ Wdi, const float* __restrict__ Wg2,
                     half_t* __restrict__ Wsw) {
  const int li = blockIdx.x * 256 + threadIdx.x;
  const int total = 3 * BIGN + S_ * DD * DD;
  if (li >= total) return;
  if (li < 3 * BIGN) {
    const int w   = li / BIGN;
    const int rem = li % BIGN;                 // [s][k][n] flat
    const int s = rem / (DIN * DD);
    const int k = (rem / DD) % DIN;
    const int n = rem % DD;
    const float* src = (w == 0) ? Wg1 : (w == 1) ? Wdr : Wdi;
    const int kc = k >> 5, kk = k & 31;
    const int nt = n >> 6, fi = (n >> 4) & 3, col = n & 15;
    const size_t dst = (size_t)w * BIGN +
        ((((size_t)(s * 4 + nt) * 32 + kc) * 4 + fi) * 32 + frag_lane(kk, col)) * FR +
        frag_elem(kk);
    Wsw[dst] = (half_t)src[rem];
  } else {
    const int rem = li - 3 * BIGN;             // [s][k][n] flat, K=256
    const int s = rem / (DD * DD);
    const int k = (rem / DD) % DD;
    const int n = rem % DD;
    const int kc = k >> 5, kk = k & 31;
    const int nt = n >> 6, fi = (n >> 4) & 3, col = n & 15;
    const size_t dst = (size_t)3 * BIGN +
        ((((size_t)(s * 4 + nt) * 8 + kc) * 4 + fi) * 32 + frag_lane(kk, col)) * FR +
        frag_elem(kk);
    Wsw[dst] = (half_t)Wg2[rem];
  }
}

// ======================= K1: fused triple GEMM (K=1024) ======================
// g1 = relu(X@Wg1+bg1) -> f16 row-major; dre/dim = X@W+b -> f32 col-major
__global__ __launch_bounds__(256)
void k1_triple_gemm(const float* __restrict__ x, const half_t* __restrict__ Wsw,
                    const float* __restrict__ bg1, const float* __restrict__ bdr,
                    const float* __restrict__ bdi,
                    half_t* __restrict__ G1h,
                    float* __restrict__ DRE, float* __restrict__ DIM) {
  __shared__ half_t Xa[4 * 32 * 32 * FR];   // 128 KB, A-fragment order

  const int tid  = threadIdx.x;
  const int lane = tid & 31;
  const int wave = tid >> 5;
  const int m_w  = wave & 3;                // 4 row sub-tiles of 16
  const int n_w  = wave >> 2;               // 2 col sub-tiles of 32
  const int row0 = blockIdx.x * MT;
  const int s    = blockIdx.y;

  // Stage X tile into A-fragment layout: float4 load -> v4h -> 8B LDS store
  for (int li = tid; li < MT * (DIN / 4); li += 256) {
    const int r  = li >> 8;                 // DIN/4 == 256
    const int k4 = (li & 255) << 2;
    const float4 v = *(const float4*)(x + (size_t)(row0 + r) * DIN + k4);
    v4h h; h.x = (half_t)v.x; h.y = (half_t)v.y; h.z = (half_t)v.z; h.w = (half_t)v.w;
    const int m_t = r >> 4, row = r & 15;
    const int kc = k4 >> 5, kk = k4 & 31;
    const int idx = (((m_t * 32 + kc) * 32 + frag_lane(kk, row)) * FR) + frag_elem(kk);
    *(v4h*)&Xa[idx] = h;
  }
  __syncthreads();

  for (int nt = 0; nt < 4; ++nt) {
    v8f acc_g[2] = {};
    v8f acc_r[2] = {};
    v8f acc_i[2] = {};

    for (int kc = 0; kc < 32; ++kc) {
      const v16h a = ld_frag(&Xa[((m_w * 32 + kc) * 32 + lane) * FR]);
#pragma unroll
      for (int f = 0; f < 2; ++f) {
        const int fi = n_w * 2 + f;
        const size_t bo =
            ((((size_t)(s * 4 + nt) * 32 + kc) * 4 + fi) * 32 + lane) * FR;
        const v16h bG = ld_frag(Wsw + 0 * (size_t)BIGN + bo);
        const v16h bR = ld_frag(Wsw + 1 * (size_t)BIGN + bo);
        const v16h bI = ld_frag(Wsw + 2 * (size_t)BIGN + bo);
        acc_g[f] = __builtin_amdgcn_wmma_f32_16x16x32_f16(
            false, a, false, bG, (short)0, acc_g[f], false, false);
        acc_r[f] = __builtin_amdgcn_wmma_f32_16x16x32_f16(
            false, a, false, bR, (short)0, acc_r[f], false, false);
        acc_i[f] = __builtin_amdgcn_wmma_f32_16x16x32_f16(
            false, a, false, bI, (short)0, acc_i[f], false, false);
      }
    }

    // Writeback
    const int mrow = row0 + m_w * 16 + ((lane >> 4) << 3);   // 8-row aligned
    const int n0   = nt * 64;
#pragma unroll
    for (int f = 0; f < 2; ++f) {
      const int col = n0 + n_w * 32 + f * 16 + (lane & 15);
      const float b1 = bg1[s * DD + col];
      const float br = bdr[s * DD + col];
      const float bi = bdi[s * DD + col];
      // g1: f16 row-major (A operand for K2)
#pragma unroll
      for (int r = 0; r < 8; ++r) {
        float g = acc_g[f][r] + b1;
        g = g > 0.f ? g : 0.f;
        G1h[(size_t)(s * NTOK + mrow + r) * DD + col] = (half_t)g;
      }
      // dre/dim: f32 col-major [s][col][row] -> two float4 per fragment
      const size_t cm = (size_t)(s * DD + col) * NTOK + mrow;
      float4 q0, q1;
      q0.x = acc_r[f][0] + br; q0.y = acc_r[f][1] + br;
      q0.z = acc_r[f][2] + br; q0.w = acc_r[f][3] + br;
      q1.x = acc_r[f][4] + br; q1.y = acc_r[f][5] + br;
      q1.z = acc_r[f][6] + br; q1.w = acc_r[f][7] + br;
      *(float4*)&DRE[cm] = q0; *(float4*)&DRE[cm + 4] = q1;
      q0.x = acc_i[f][0] + bi; q0.y = acc_i[f][1] + bi;
      q0.z = acc_i[f][2] + bi; q0.w = acc_i[f][3] + bi;
      q1.x = acc_i[f][4] + bi; q1.y = acc_i[f][5] + bi;
      q1.z = acc_i[f][6] + bi; q1.w = acc_i[f][7] + bi;
      *(float4*)&DIM[cm] = q0; *(float4*)&DIM[cm + 4] = q1;
    }
  }
}

// ========== K2: gate GEMM (K=256) + sigmoid + elementwise a/b fields =========
__global__ __launch_bounds__(256)
void k2_gate_ew(const half_t* __restrict__ G1h, const half_t* __restrict__ Wsw,
                const float* __restrict__ bg2,
                const float* __restrict__ DRE, const float* __restrict__ DIM,
                float* __restrict__ Are, float* __restrict__ Aim,
                float* __restrict__ out) {
  const int tid  = threadIdx.x;
  const int lane = tid & 31;
  const int wave = tid >> 5;
  const int m_w  = wave & 3;
  const int n_w  = wave >> 2;
  const int row0 = blockIdx.x * MT;
  const int s    = blockIdx.y;

  const float are_b = c_r[s] * __cosf(c_th[s]);
  const float aim_b = c_r[s] * __sinf(c_th[s]);
  const float bs    = c_bs[s];

  // A fragment base for this wave: row = lane&15 within m-tile, kg per lane half
  const half_t* abase =
      G1h + (size_t)(s * NTOK + row0 + m_w * 16 + (lane & 15)) * DD +
      ((lane >> 4) << 3);

  for (int nt = 0; nt < 4; ++nt) {
    v8f acc[2] = {};

    for (int kc = 0; kc < 8; ++kc) {
      const half_t* ap = abase + kc * 32;
      const v8h lo = *(const v8h*)ap;          // K kg..kg+7
      const v8h hi = *(const v8h*)(ap + 16);   // K 16+kg..16+kg+7
      const v16h a = __builtin_shufflevector(lo, hi, 0, 1, 2, 3, 4, 5, 6, 7,
                                             8, 9, 10, 11, 12, 13, 14, 15);
#pragma unroll
      for (int f = 0; f < 2; ++f) {
        const int fi = n_w * 2 + f;
        const size_t bo = (size_t)3 * BIGN +
            ((((size_t)(s * 4 + nt) * 8 + kc) * 4 + fi) * 32 + lane) * FR;
        const v16h b = ld_frag(Wsw + bo);
        acc[f] = __builtin_amdgcn_wmma_f32_16x16x32_f16(
            false, a, false, b, (short)0, acc[f], false, false);
      }
    }

    const int mrow = row0 + m_w * 16 + ((lane >> 4) << 3);
    const int n0   = nt * 64;
#pragma unroll
    for (int f = 0; f < 2; ++f) {
      const int col = n0 + n_w * 32 + f * 16 + (lane & 15);
      const float b2 = bg2[s * DD + col];
      const size_t cm = (size_t)(s * DD + col) * NTOK + mrow;  // col-major dre/dim
      const float4 dr0 = *(const float4*)&DRE[cm];
      const float4 dr1 = *(const float4*)&DRE[cm + 4];
      const float4 di0 = *(const float4*)&DIM[cm];
      const float4 di1 = *(const float4*)&DIM[cm + 4];
      const float dre[8] = {dr0.x, dr0.y, dr0.z, dr0.w, dr1.x, dr1.y, dr1.z, dr1.w};
      const float dim[8] = {di0.x, di0.y, di0.z, di0.w, di1.x, di1.y, di1.z, di1.w};
#pragma unroll
      for (int r = 0; r < 8; ++r) {
        const int row = mrow + r;
        const float gate = 1.f / (1.f + __expf(-(acc[f][r] + b2)));
        const float omg  = 1.f - gate;
        const size_t ia = (size_t)(s * NTOK + row) * DD + col;
        Are[ia] = omg * are_b;
        Aim[ia] = omg * aim_b;
        const size_t o = ((size_t)(row * S_ + s) * 2) * DD + col;
        out[o]      = omg * dre[r] * bs;   // b real
        out[o + DD] = omg * dim[r] * bs;   // b imag
      }
    }
  }
}

// ============== K3: per-chunk complex prefix scan, in place ==================
__global__ __launch_bounds__(256)
void k3_chunk_scan(float* __restrict__ Are, float* __restrict__ Aim,
                   float* __restrict__ out) {
  const int linear = blockIdx.x * 256 + threadIdx.x;  // S*B*NCHUNK*DD threads
  const int d    = linear & (DD - 1);
  const int rest = linear >> 8;
  const int c    = rest & (NCHUNK - 1);
  const int sb   = rest >> 6;
  const int b    = sb & 3;
  const int s    = sb >> 2;

  float pAr = 1.f, pAi = 0.f, pBr = 0.f, pBi = 0.f;
  const int t0 = c * CHUNK;
  for (int t = t0; t < t0 + CHUNK; ++t) {
    const int row = b * T_ + t;
    const size_t ia = (size_t)(s * NTOK + row) * DD + d;
    const size_t ob = ((size_t)(row * S_ + s) * 2) * DD + d;
    const float ar = Are[ia], ai = Aim[ia];
    const float br = out[ob], bi = out[ob + DD];
    const float nAr = ar * pAr - ai * pAi;
    const float nAi = ar * pAi + ai * pAr;
    const float nBr = ar * pBr - ai * pBi + br;
    const float nBi = ar * pBi + ai * pBr + bi;
    pAr = nAr; pAi = nAi; pBr = nBr; pBi = nBi;
    Are[ia] = pAr; Aim[ia] = pAi;
    out[ob] = pBr; out[ob + DD] = pBi;
  }
}

// ============== K4: carry scan across chunk aggregates =======================
__global__ __launch_bounds__(256)
void k4_carry(const float* __restrict__ Are, const float* __restrict__ Aim,
              const float* __restrict__ out,
              float* __restrict__ Bcr, float* __restrict__ Bci) {
  const int linear = blockIdx.x * 256 + threadIdx.x;  // S*B*DD threads
  const int d  = linear & (DD - 1);
  const int sb = linear >> 8;
  const int b  = sb & 3;
  const int s  = sb >> 2;

  float cAr = 1.f, cAi = 0.f, cBr = 0.f, cBi = 0.f;
  for (int c = 0; c < NCHUNK; ++c) {
    const size_t ic = ((size_t)(sb * NCHUNK + c)) * DD + d;
    Bcr[ic] = cBr; Bci[ic] = cBi;
    const int t   = c * CHUNK + CHUNK - 1;
    const int row = b * T_ + t;
    const size_t ia = (size_t)(s * NTOK + row) * DD + d;
    const size_t ob = ((size_t)(row * S_ + s) * 2) * DD + d;
    const float aAr = Are[ia], aAi = Aim[ia];
    const float aBr = out[ob], aBi = out[ob + DD];
    const float nAr = aAr * cAr - aAi * cAi;
    const float nAi = aAr * cAi + aAi * cAr;
    const float nBr = aAr * cBr - aAi * cBi + aBr;
    const float nBi = aAr * cBi + aAi * cBr + aBi;
    cAr = nAr; cAi = nAi; cBr = nBr; cBi = nBi;
  }
}

// ============== K5: parallel apply h = A_loc*B_carry + B_loc =================
__global__ __launch_bounds__(256)
void k5_apply(const float* __restrict__ Are, const float* __restrict__ Aim,
              const float* __restrict__ Bcr, const float* __restrict__ Bci,
              float* __restrict__ out) {
  const size_t linear = (size_t)blockIdx.x * 256 + threadIdx.x;  // S*NTOK*DD
  const int d    = (int)(linear & (DD - 1));
  const size_t rest = linear >> 8;            // s*NTOK + row
  const int row  = (int)(rest & (NTOK - 1));
  const int s    = (int)(rest >> 14);
  const int b    = row >> 12;
  const int t    = row & (T_ - 1);
  const int c    = t / CHUNK;
  const int sb   = s * 4 + b;

  const size_t ic = ((size_t)(sb * NCHUNK + c)) * DD + d;
  const size_t ia = linear;
  const size_t ob = ((size_t)(row * S_ + s) * 2) * DD + d;

  const float Ar = Are[ia], Ai = Aim[ia];
  const float Br = out[ob], Bi = out[ob + DD];
  const float br = Bcr[ic], bi = Bci[ic];
  out[ob]      = Ar * br - Ai * bi + Br;
  out[ob + DD] = Ar * bi + Ai * br + Bi;
}

// ============================== launcher =====================================
extern "C" void kernel_launch(void* const* d_in, const int* in_sizes, int n_in,
                              void* d_out, int out_size, void* d_ws, size_t ws_size,
                              hipStream_t stream) {
  const float* x   = (const float*)d_in[0];
  const float* Wg1 = (const float*)d_in[1];
  const float* bg1 = (const float*)d_in[2];
  const float* Wg2 = (const float*)d_in[3];
  const float* bg2 = (const float*)d_in[4];
  const float* Wdr = (const float*)d_in[5];
  const float* bdr = (const float*)d_in[6];
  const float* Wdi = (const float*)d_in[7];
  const float* bdi = (const float*)d_in[8];
  float* out = (float*)d_out;

  const size_t nField = (size_t)S_ * NTOK * DD;   // 16,777,216
  char* ws = (char*)d_ws;
  float*  Are = (float*)ws;  ws += nField * sizeof(float);
  float*  Aim = (float*)ws;  ws += nField * sizeof(float);
  float*  DRE = (float*)ws;  ws += nField * sizeof(float);
  float*  DIM = (float*)ws;  ws += nField * sizeof(float);
  half_t* G1h = (half_t*)ws; ws += nField * sizeof(half_t);
  float*  Bcr = (float*)ws;  ws += (size_t)S_ * B_ * NCHUNK * DD * sizeof(float);
  float*  Bci = (float*)ws;  ws += (size_t)S_ * B_ * NCHUNK * DD * sizeof(float);
  half_t* Wsw = (half_t*)ws;

  // K0: weight pre-swizzle (fp32 -> f16, WMMA B-fragment order)
  const int wtotal = 3 * BIGN + S_ * DD * DD;
  k0_prep_weights<<<(wtotal + 255) / 256, 256, 0, stream>>>(Wg1, Wdr, Wdi, Wg2, Wsw);

  // K1: triple GEMM
  k1_triple_gemm<<<dim3(NTOK / MT, S_), 256, 0, stream>>>(
      x, Wsw, bg1, bdr, bdi, G1h, DRE, DIM);

  // K2: gate GEMM + elementwise a/b
  k2_gate_ew<<<dim3(NTOK / MT, S_), 256, 0, stream>>>(
      G1h, Wsw, bg2, DRE, DIM, Are, Aim, out);

  // K3: per-chunk prefix scan
  k3_chunk_scan<<<(S_ * B_ * NCHUNK * DD) / 256, 256, 0, stream>>>(Are, Aim, out);

  // K4: carry scan across chunks
  k4_carry<<<(S_ * B_ * DD) / 256, 256, 0, stream>>>(Are, Aim, out, Bcr, Bci);

  // K5: parallel apply
  k5_apply<<<(int)(nField / 256), 256, 0, stream>>>(Are, Aim, Bcr, Bci, out);
}
